// DeepseekV2DecoderLayer_13907104105111
// MI455X (gfx1250) — matrix operand order repaired and checked by reading, not compiled, not run on hardware
//
#include <hip/hip_runtime.h>
#include <hip/hip_bf16.h>

// ---------------------------------------------------------------------------
// DeepSeek-V2 MLA decoder layer for gfx1250 (MI455X).
// bf16 WMMA for every matmul; async global->LDS DMA + double buffering in the
// GEMMs; flash-style WMMA attention with software-pipelined fragment loads.
// ---------------------------------------------------------------------------

typedef __attribute__((ext_vector_type(16))) __bf16 v16bf;
typedef __attribute__((ext_vector_type(8)))  float  v8f;
typedef __attribute__((ext_vector_type(4)))  int    v4i;

#define D_B    2
#define D_S    2048
#define D_H    2048
#define D_NH   16
#define D_QLR  1536
#define D_KVLR 512
#define D_ROPE 64
#define D_NOPE 128
#define D_VD   128
#define D_QHD  192            // NOPE + ROPE
#define D_IM   10944
#define D_TOK  (D_B * D_S)    // 4096 tokens
#define ATT_SCALE 0.07216878364870322f   // 192^-0.5

__device__ __forceinline__ unsigned short f32_bf16(float f) {
  union { float f; unsigned u; } v; v.f = f;
  unsigned r = v.u + 0x7fffu + ((v.u >> 16) & 1u);   // round-to-nearest-even
  return (unsigned short)(r >> 16);
}

union FragA { v16bf v; unsigned u[8]; unsigned short h[16]; };
union FragC { v8f   v; float f[8]; };

// ---------------------------------------------------------------------------
// CDNA5 async global->LDS (ASYNCcnt-tracked DMA), guarded so a missing builtin
// falls back to a synchronous copy without breaking the build.
// Builtin signature (per clang diagnostic): (int4 AS1*, int4 AS3*, Ii, Ii).
// ---------------------------------------------------------------------------
#if __has_builtin(__builtin_amdgcn_global_load_async_to_lds_b128)
#define HAVE_ASYNC_LDS 1
#else
#define HAVE_ASYNC_LDS 0
#endif

typedef __attribute__((address_space(1))) v4i* as1v4p;
typedef __attribute__((address_space(3))) v4i* as3v4p;

__device__ __forceinline__ void cp16_to_lds(void* lds, const void* g) {
#if HAVE_ASYNC_LDS
  __builtin_amdgcn_global_load_async_to_lds_b128((as1v4p)g, (as3v4p)lds, 0, 0);
#else
  *(uint4*)lds = *(const uint4*)g;
#endif
}

__device__ __forceinline__ void wait_async_lds() {
#if HAVE_ASYNC_LDS
#if __has_builtin(__builtin_amdgcn_s_wait_asynccnt)
  __builtin_amdgcn_s_wait_asynccnt(0);
#else
  asm volatile("s_wait_asynccnt 0" ::: "memory");
#endif
#endif
}

// ---------------------------------------------------------------------------
// Generic bf16 WMMA GEMM: C[M,N] = A[M,K] * B[K,N], f32 accumulate.
// Block tile 128x64, BK=32; 8 waves, each wave computes 32x32 (2x2 WMMA tiles).
// Double-buffered LDS filled by async global->LDS b128 DMA.
// EPI: 0 = f32 store, 1 = bf16 store, 2 = bf16(silu(acc)*aux), 3 = f32(acc+aux)
// ---------------------------------------------------------------------------
#define BM 128
#define BN 64
#define BK 32
#define LDA (BK + 8)   // 40 halves -> 80 B row stride (16B-aligned)
#define LDB (BN + 8)   // 72 halves -> 144 B row stride (16B-aligned)

template <int EPI>
__global__ __launch_bounds__(256)
void wmma_gemm_kernel(const unsigned short* __restrict__ A,
                      const unsigned short* __restrict__ B,
                      const float* __restrict__ aux,
                      void* __restrict__ Cvp,
                      int M, int N, int K)
{
  __shared__ __align__(16) unsigned short As[2][BM * LDA];
  __shared__ __align__(16) unsigned short Bs[2][BK * LDB];

  const int tid  = threadIdx.x;
  const int lane = tid & 31;
  const int wave = tid >> 5;
  const int wm   = wave >> 1;       // 0..3 (M dir)
  const int wn   = wave & 1;        // 0..1 (N dir)
  const int lm   = lane & 15;
  const int lh   = lane >> 4;       // half-wave select

  const int bn0 = blockIdx.x * BN;
  const int bm0 = blockIdx.y * BM;

  // async-stage one (BM x BK) A tile + (BK x BN) B tile in 16-byte chunks
  auto stage = [&](int buf, int k0) {
#pragma unroll
    for (int it = 0; it < 2; ++it) {            // A: 512 x 16B chunks
      const int i = tid + it * 256;
      const int r = i >> 2, c = i & 3;
      cp16_to_lds(&As[buf][r * LDA + c * 8],
                  A + (size_t)(bm0 + r) * K + k0 + c * 8);
    }
    {                                           // B: 256 x 16B chunks
      const int r = tid >> 3, c = tid & 7;
      cp16_to_lds(&Bs[buf][r * LDB + c * 8],
                  B + (size_t)(k0 + r) * N + bn0 + c * 8);
    }
  };

  FragC acc[2][2];
#pragma unroll
  for (int i = 0; i < 2; ++i)
#pragma unroll
    for (int j = 0; j < 2; ++j)
#pragma unroll
      for (int r = 0; r < 8; ++r) acc[i][j].f[r] = 0.0f;

  stage(0, 0);
  wait_async_lds();
  __syncthreads();

  int buf = 0;
  for (int k0 = 0; k0 < K; k0 += BK) {
    if (k0 + BK < K) stage(buf ^ 1, k0 + BK);   // DMA next tile while computing

    // ---- fragment loads per ISA 7.12.2 layouts ----
    FragA af[2], bfrag[2];
#pragma unroll
    for (int mt = 0; mt < 2; ++mt) {
      const unsigned short* ar = &As[buf][(wm * 32 + mt * 16 + lm) * LDA];
#pragma unroll
      for (int v = 0; v < 4; ++v) {
        af[mt].u[v]     = *(const unsigned*)&ar[lh * 8 + 2 * v];        // K 0..15 half
        af[mt].u[4 + v] = *(const unsigned*)&ar[16 + lh * 8 + 2 * v];   // K 16..31 half
      }
    }
#pragma unroll
    for (int nt = 0; nt < 2; ++nt) {
      const int n = wn * 32 + nt * 16 + lm;
#pragma unroll
      for (int v = 0; v < 8; ++v) {
        const int k = lh * 8 + 2 * v + (v >= 4 ? 8 : 0);   // B: K = 8*lh + s (s<8), 16 + 8*lh + (s-8) (s>=8)
        bfrag[nt].h[2 * v]     = Bs[buf][k * LDB + n];
        bfrag[nt].h[2 * v + 1] = Bs[buf][(k + 1) * LDB + n];
      }
    }
#pragma unroll
    for (int mt = 0; mt < 2; ++mt)
#pragma unroll
      for (int nt = 0; nt < 2; ++nt)
        acc[mt][nt].v = __builtin_amdgcn_wmma_f32_16x16x32_bf16(
            false, af[mt].v, false, bfrag[nt].v, (short)0, acc[mt][nt].v,
            false, false);

    wait_async_lds();      // our wave's DMA into buf^1 done
    __syncthreads();       // everyone's DMA done + reads of buf finished
    buf ^= 1;
  }

  // ---- epilogue (C layout: row = r + lh*8, col = lm) ----
#pragma unroll
  for (int mt = 0; mt < 2; ++mt)
#pragma unroll
    for (int nt = 0; nt < 2; ++nt)
#pragma unroll
      for (int r = 0; r < 8; ++r) {
        const int row = bm0 + wm * 32 + mt * 16 + r + lh * 8;
        const int col = bn0 + wn * 32 + nt * 16 + lm;
        const size_t idx = (size_t)row * N + col;
        const float v = acc[mt][nt].f[r];
        if (EPI == 0) {
          ((float*)Cvp)[idx] = v;
        } else if (EPI == 1) {
          ((unsigned short*)Cvp)[idx] = f32_bf16(v);
        } else if (EPI == 2) {
          const float s = v / (1.0f + __expf(-v));     // silu
          ((unsigned short*)Cvp)[idx] = f32_bf16(s * aux[idx]);
        } else {
          ((float*)Cvp)[idx] = v + aux[idx];           // residual add
        }
      }
}

// ---------------------------------------------------------------------------
// RMSNorm (f32 in, bf16*weight out). One block per row.
// ---------------------------------------------------------------------------
__global__ __launch_bounds__(256)
void rmsnorm_kernel(const float* __restrict__ x, const float* __restrict__ w,
                    unsigned short* __restrict__ out, int width, int stride)
{
  const int row = blockIdx.x;
  const float* xr = x + (size_t)row * stride;
  float s = 0.f;
  for (int i = threadIdx.x; i < width; i += 256) { const float v = xr[i]; s += v * v; }
#pragma unroll
  for (int off = 16; off > 0; off >>= 1) s += __shfl_xor(s, off, 32);
  __shared__ float red[8];
  __shared__ float invs;
  if ((threadIdx.x & 31) == 0) red[threadIdx.x >> 5] = s;
  __syncthreads();
  if (threadIdx.x == 0) {
    float t = 0.f;
    for (int i = 0; i < 8; ++i) t += red[i];
    invs = rsqrtf(t / (float)width + 1e-6f);
  }
  __syncthreads();
  const float r = invs;
  unsigned short* orow = out + (size_t)row * width;
  for (int i = threadIdx.x; i < width; i += 256)
    orow[i] = f32_bf16(xr[i] * r * w[i]);
}

// ---------------------------------------------------------------------------
// fp32 -> bf16 cast (weights / activations)
// ---------------------------------------------------------------------------
__global__ void cast_bf16_kernel(const float* __restrict__ in,
                                 unsigned short* __restrict__ out, size_t n)
{
  size_t i = (size_t)blockIdx.x * blockDim.x + threadIdx.x;
  const size_t st = (size_t)gridDim.x * blockDim.x;
  for (; i < n; i += st) out[i] = f32_bf16(in[i]);
}

// ---------------------------------------------------------------------------
// RoPE (deinterleaved) + assembly of qs/ks/v into [B,NH,S,D] bf16 layouts.
// q is pre-scaled by 192^-0.5. One block per (b,s) token.
// ---------------------------------------------------------------------------
__global__ __launch_bounds__(128)
void rope_assemble_kernel(const float* __restrict__ q, const float* __restrict__ kv,
                          const float* __restrict__ ckv, const float* __restrict__ cosT,
                          const float* __restrict__ sinT, const int* __restrict__ pos_ids,
                          unsigned short* __restrict__ qs, unsigned short* __restrict__ ks,
                          unsigned short* __restrict__ vb)
{
  const int row = blockIdx.x;          // b*S + s
  const int b = row >> 11;
  const int s = row & (D_S - 1);
  const int pos = pos_ids[row];
  const float* cs = cosT + (size_t)pos * D_ROPE;
  const float* sn = sinT + (size_t)pos * D_ROPE;

  __shared__ float kpe[D_ROPE];
  if (threadIdx.x < D_ROPE) {
    const int j = threadIdx.x;
    const float* kp = ckv + (size_t)row * (D_KVLR + D_ROPE) + D_KVLR;
    float v;
    if (j < 32) v = kp[2 * j] * cs[j] - kp[2 * j + 1] * sn[j];
    else { const int jj = j - 32; v = kp[2 * jj + 1] * cs[j] + kp[2 * jj] * sn[j]; }
    kpe[j] = v;
  }
  __syncthreads();

  const float* qrow  = q  + (size_t)row * (D_NH * D_QHD);
  const float* kvrow = kv + (size_t)row * (D_NH * (D_NOPE + D_VD));

  for (int i = threadIdx.x; i < D_NH * D_QHD; i += blockDim.x) {
    const int hh = i / D_QHD, d = i % D_QHD;
    const size_t dst = (((size_t)(b * D_NH + hh)) * D_S + s) * D_QHD + d;
    const float* qh = qrow + hh * D_QHD;
    float qv;
    if (d < D_NOPE) qv = qh[d];
    else {
      const int j = d - D_NOPE;
      if (j < 32) qv = qh[D_NOPE + 2 * j] * cs[j] - qh[D_NOPE + 2 * j + 1] * sn[j];
      else { const int jj = j - 32; qv = qh[D_NOPE + 2 * jj + 1] * cs[j] + qh[D_NOPE + 2 * jj] * sn[j]; }
    }
    qs[dst] = f32_bf16(qv * ATT_SCALE);
    const float kval = (d < D_NOPE) ? kvrow[hh * (D_NOPE + D_VD) + d] : kpe[d - D_NOPE];
    ks[dst] = f32_bf16(kval);
  }
  for (int i = threadIdx.x; i < D_NH * D_VD; i += blockDim.x) {
    const int hh = i >> 7, d = i & (D_VD - 1);
    vb[(((size_t)(b * D_NH + hh)) * D_S + s) * D_VD + d] =
        f32_bf16(kvrow[hh * (D_NOPE + D_VD) + D_NOPE + d]);
  }
}

// ---------------------------------------------------------------------------
// Flash-style causal MLA attention. One wave (32 threads) per 16 query rows
// of one (b,h). Q held as 6 resident A-fragments (16x192). Per 32-key block:
// 12 WMMAs for QK^T, online softmax with half-wave shfl reductions, P via LDS
// re-layout, 8 WMMAs for P*V. Fragment loads are software-pipelined (ping-pong
// buffers) so the next load is in flight while the current WMMA executes.
// Output written as bf16 [B, S, NH*VD] for the o_proj GEMM.
// ---------------------------------------------------------------------------
__global__ __launch_bounds__(32)
void mla_attn_kernel(const unsigned short* __restrict__ qs,
                     const unsigned short* __restrict__ ks,
                     const unsigned short* __restrict__ vb,
                     unsigned short* __restrict__ attn)
{
  const int qt = blockIdx.x & (D_S / 16 - 1);
  const int bh = blockIdx.x >> 7;                 // (b*NH + h)
  const int b  = bh >> 4, h = bh & (D_NH - 1);
  const int q0 = qt * 16;
  const int lane = threadIdx.x;
  const int lm = lane & 15, lh = lane >> 4;

  const unsigned short* Q  = qs + (size_t)bh * D_S * D_QHD;
  const unsigned short* Kp = ks + (size_t)bh * D_S * D_QHD;
  const unsigned short* Vp = vb + (size_t)bh * D_S * D_VD;

  // Q A-fragments: 6 chunks of K=32 over QHD=192
  FragA qf[6];
  {
    const unsigned* qr = (const unsigned*)(Q + (size_t)(q0 + lm) * D_QHD);
#pragma unroll
    for (int c = 0; c < 6; ++c)
#pragma unroll
      for (int v = 0; v < 4; ++v) {
        qf[c].u[v]     = qr[c * 16 + lh * 4 + v];
        qf[c].u[4 + v] = qr[c * 16 + 8 + lh * 4 + v];
      }
  }

  FragC o[8];
#pragma unroll
  for (int nt = 0; nt < 8; ++nt)
#pragma unroll
    for (int r = 0; r < 8; ++r) o[nt].f[r] = 0.0f;
  float rmax[8], rsum[8];
#pragma unroll
  for (int r = 0; r < 8; ++r) { rmax[r] = -1e30f; rsum[r] = 0.0f; }

  __shared__ __align__(16) unsigned short Pl[16 * 40];   // 16 x (32+8) bf16 P tile

  const int kend = q0 + 16;                // keys <= q0+15 can be unmasked
  for (int kb = 0; kb < kend; kb += 32) {
    // ---- scores: two 16x16 tiles over keys kb..kb+31 (pipelined loads) ----
    FragC sc[2];
#pragma unroll
    for (int t = 0; t < 2; ++t) {
#pragma unroll
      for (int r = 0; r < 8; ++r) sc[t].f[r] = 0.0f;
      const unsigned* kr = (const unsigned*)(Kp + (size_t)(kb + t * 16 + lm) * D_QHD);
      FragA bf[2];
#pragma unroll
      for (int v = 0; v < 8; ++v) bf[0].u[v] = kr[lh * 4 + v + (v >= 4 ? 4 : 0)];
#pragma unroll
      for (int c = 0; c < 6; ++c) {
        if (c < 5) {
#pragma unroll
          for (int v = 0; v < 8; ++v)
            bf[(c + 1) & 1].u[v] = kr[(c + 1) * 16 + lh * 4 + v + (v >= 4 ? 4 : 0)];
        }
        sc[t].v = __builtin_amdgcn_wmma_f32_16x16x32_bf16(
            false, qf[c].v, false, bf[c & 1].v, (short)0, sc[t].v, false, false);
      }
    }

    // ---- causal mask + online softmax (row = r + lh*8, key = kb + t*16 + lm) ----
#pragma unroll
    for (int r = 0; r < 8; ++r) {
      const int qrow = q0 + r + lh * 8;
#pragma unroll
      for (int t = 0; t < 2; ++t)
        if (kb + t * 16 + lm > qrow) sc[t].f[r] = -1e30f;
      float mx = fmaxf(sc[0].f[r], sc[1].f[r]);
#pragma unroll
      for (int off = 1; off < 16; off <<= 1) mx = fmaxf(mx, __shfl_xor(mx, off, 32));
      const float nm = fmaxf(rmax[r], mx);
      const float corr = __expf(rmax[r] - nm);
      rsum[r] *= corr;
#pragma unroll
      for (int nt = 0; nt < 8; ++nt) o[nt].f[r] *= corr;
      float ps = 0.f;
#pragma unroll
      for (int t = 0; t < 2; ++t) {
        const float p = __expf(sc[t].f[r] - nm);
        sc[t].f[r] = p; ps += p;
      }
#pragma unroll
      for (int off = 1; off < 16; off <<= 1) ps += __shfl_xor(ps, off, 32);
      rsum[r] += ps;
      rmax[r] = nm;
    }

    // ---- P: C-layout -> LDS -> A-fragment (16x32) ----
#pragma unroll
    for (int r = 0; r < 8; ++r) {
      Pl[(r + lh * 8) * 40 + lm]      = f32_bf16(sc[0].f[r]);
      Pl[(r + lh * 8) * 40 + 16 + lm] = f32_bf16(sc[1].f[r]);
    }
    __syncthreads();
    FragA pf;
    {
      const unsigned* pr = (const unsigned*)&Pl[lm * 40];
#pragma unroll
      for (int v = 0; v < 4; ++v) {
        pf.u[v]     = pr[lh * 4 + v];
        pf.u[4 + v] = pr[8 + lh * 4 + v];
      }
    }
    // ---- P * V : 8 N-tiles of 16 over VD=128 (pipelined V loads) ----
    FragA vf[2];
#pragma unroll
    for (int v = 0; v < 8; ++v) {
      const int k = lh * 8 + 2 * v + (v >= 4 ? 8 : 0);
      vf[0].h[2 * v]     = Vp[(size_t)(kb + k) * D_VD + lm];
      vf[0].h[2 * v + 1] = Vp[(size_t)(kb + k + 1) * D_VD + lm];
    }
#pragma unroll
    for (int nt = 0; nt < 8; ++nt) {
      if (nt < 7) {
#pragma unroll
        for (int v = 0; v < 8; ++v) {
          const int k = lh * 8 + 2 * v + (v >= 4 ? 8 : 0);
          vf[(nt + 1) & 1].h[2 * v]     = Vp[(size_t)(kb + k) * D_VD + (nt + 1) * 16 + lm];
          vf[(nt + 1) & 1].h[2 * v + 1] = Vp[(size_t)(kb + k + 1) * D_VD + (nt + 1) * 16 + lm];
        }
      }
      o[nt].v = __builtin_amdgcn_wmma_f32_16x16x32_bf16(
          false, pf.v, false, vf[nt & 1].v, (short)0, o[nt].v, false, false);
    }
    __syncthreads();
  }

  // ---- finalize + store bf16 [B, S, NH*VD] ----
#pragma unroll
  for (int nt = 0; nt < 8; ++nt)
#pragma unroll
    for (int r = 0; r < 8; ++r) {
      const int qrow = q0 + r + lh * 8;
      const float val = o[nt].f[r] / rsum[r];
      attn[((size_t)(b * D_S + qrow)) * (D_NH * D_VD) + h * D_VD + nt * 16 + lm] =
          f32_bf16(val);
    }
}

// ---------------------------------------------------------------------------
// Host-side orchestration
// ---------------------------------------------------------------------------
extern "C" void kernel_launch(void* const* d_in, const int* in_sizes, int n_in,
                              void* d_out, int out_size, void* d_ws, size_t ws_size,
                              hipStream_t stream)
{
  (void)in_sizes; (void)n_in; (void)out_size; (void)ws_size;

  const float* hidden  = (const float*)d_in[0];
  const float* q_a_w   = (const float*)d_in[1];
  const float* q_a_ln  = (const float*)d_in[2];
  const float* q_b_w   = (const float*)d_in[3];
  const float* kv_a_w  = (const float*)d_in[4];
  const float* kv_a_ln = (const float*)d_in[5];
  const float* kv_b_w  = (const float*)d_in[6];
  const float* o_w     = (const float*)d_in[7];
  const float* in_ln   = (const float*)d_in[8];
  const float* post_ln = (const float*)d_in[9];
  const float* gate_w  = (const float*)d_in[10];
  const float* up_w    = (const float*)d_in[11];
  const float* down_w  = (const float*)d_in[12];
  const float* cosT    = (const float*)d_in[13];
  const float* sinT    = (const float*)d_in[14];
  const int*   pos_ids = (const int*)d_in[15];

  char* base = (char*)d_ws;
  size_t off = 0;
  auto carve = [&](size_t bytes) -> void* {
    void* r = base + off;
    off = (off + bytes + 255) & ~(size_t)255;
    return r;
  };

  // bf16 weight copies
  unsigned short* w_qa   = (unsigned short*)carve((size_t)D_H * D_QLR * 2);
  unsigned short* w_qb   = (unsigned short*)carve((size_t)D_QLR * (D_NH * D_QHD) * 2);
  unsigned short* w_kva  = (unsigned short*)carve((size_t)D_H * (D_KVLR + D_ROPE) * 2);
  unsigned short* w_kvb  = (unsigned short*)carve((size_t)D_KVLR * (D_NH * (D_NOPE + D_VD)) * 2);
  unsigned short* w_o    = (unsigned short*)carve((size_t)(D_NH * D_VD) * D_H * 2);
  unsigned short* w_gate = (unsigned short*)carve((size_t)D_H * D_IM * 2);
  unsigned short* w_up   = (unsigned short*)carve((size_t)D_H * D_IM * 2);
  unsigned short* w_down = (unsigned short*)carve((size_t)D_IM * D_H * 2);
  // activations
  unsigned short* xb    = (unsigned short*)carve((size_t)D_TOK * D_H * 2);
  float*          qa_f  = (float*)carve((size_t)D_TOK * D_QLR * 4);
  unsigned short* qa_n  = (unsigned short*)carve((size_t)D_TOK * D_QLR * 2);
  float*          q_f   = (float*)carve((size_t)D_TOK * (D_NH * D_QHD) * 4);
  float*          ckv_f = (float*)carve((size_t)D_TOK * (D_KVLR + D_ROPE) * 4);
  unsigned short* ckv_n = (unsigned short*)carve((size_t)D_TOK * D_KVLR * 2);
  float*          kv_f  = (float*)carve((size_t)D_TOK * (D_NH * (D_NOPE + D_VD)) * 4);
  unsigned short* qsb   = (unsigned short*)carve((size_t)D_TOK * (D_NH * D_QHD) * 2);
  unsigned short* ksb   = (unsigned short*)carve((size_t)D_TOK * (D_NH * D_QHD) * 2);
  unsigned short* vbb   = (unsigned short*)carve((size_t)D_TOK * (D_NH * D_VD) * 2);
  unsigned short* attb  = (unsigned short*)carve((size_t)D_TOK * (D_NH * D_VD) * 2);
  float*          h1    = (float*)carve((size_t)D_TOK * D_H * 4);
  unsigned short* yb    = (unsigned short*)carve((size_t)D_TOK * D_H * 2);
  float*          up_f  = (float*)carve((size_t)D_TOK * D_IM * 4);
  unsigned short* actb  = (unsigned short*)carve((size_t)D_TOK * D_IM * 2);

  // 1) cast weights to bf16
  struct { const float* s; unsigned short* d; size_t n; } casts[] = {
    { q_a_w,  w_qa,   (size_t)D_H * D_QLR },
    { q_b_w,  w_qb,   (size_t)D_QLR * (D_NH * D_QHD) },
    { kv_a_w, w_kva,  (size_t)D_H * (D_KVLR + D_ROPE) },
    { kv_b_w, w_kvb,  (size_t)D_KVLR * (D_NH * (D_NOPE + D_VD)) },
    { o_w,    w_o,    (size_t)(D_NH * D_VD) * D_H },
    { gate_w, w_gate, (size_t)D_H * D_IM },
    { up_w,   w_up,   (size_t)D_H * D_IM },
    { down_w, w_down, (size_t)D_IM * D_H },
  };
  for (auto& c : casts)
    cast_bf16_kernel<<<4096, 256, 0, stream>>>(c.s, c.d, c.n);

  const dim3 blk(256);
  const int MG = D_TOK / BM;   // 32

  // 2) input RMSNorm -> bf16
  rmsnorm_kernel<<<D_TOK, blk, 0, stream>>>(hidden, in_ln, xb, D_H, D_H);

  // 3) q_a = xb @ Wqa  (f32)
  wmma_gemm_kernel<0><<<dim3(D_QLR / BN, MG), blk, 0, stream>>>(
      xb, w_qa, nullptr, qa_f, D_TOK, D_QLR, D_H);
  // 4) q_a RMSNorm -> bf16
  rmsnorm_kernel<<<D_TOK, blk, 0, stream>>>(qa_f, q_a_ln, qa_n, D_QLR, D_QLR);
  // 5) q = qa_n @ Wqb  (f32)
  wmma_gemm_kernel<0><<<dim3((D_NH * D_QHD) / BN, MG), blk, 0, stream>>>(
      qa_n, w_qb, nullptr, q_f, D_TOK, D_NH * D_QHD, D_QLR);
  // 6) ckv = xb @ Wkva (f32, 576 cols)
  wmma_gemm_kernel<0><<<dim3((D_KVLR + D_ROPE) / BN, MG), blk, 0, stream>>>(
      xb, w_kva, nullptr, ckv_f, D_TOK, D_KVLR + D_ROPE, D_H);
  // 7) compressed-kv RMSNorm (first 512 cols) -> bf16
  rmsnorm_kernel<<<D_TOK, blk, 0, stream>>>(ckv_f, kv_a_ln, ckv_n, D_KVLR, D_KVLR + D_ROPE);
  // 8) kv = ckv_n @ Wkvb (f32)
  wmma_gemm_kernel<0><<<dim3((D_NH * (D_NOPE + D_VD)) / BN, MG), blk, 0, stream>>>(
      ckv_n, w_kvb, nullptr, kv_f, D_TOK, D_NH * (D_NOPE + D_VD), D_KVLR);

  // 9) RoPE + assemble qs/ks/v bf16 head layouts
  rope_assemble_kernel<<<D_TOK, 128, 0, stream>>>(
      q_f, kv_f, ckv_f, cosT, sinT, pos_ids, qsb, ksb, vbb);

  // 10) flash MLA attention (WMMA QK^T and PV)
  mla_attn_kernel<<<D_B * D_NH * (D_S / 16), 32, 0, stream>>>(qsb, ksb, vbb, attb);

  // 11) h1 = attn @ Wo + hidden (residual fused)
  wmma_gemm_kernel<3><<<dim3(D_H / BN, MG), blk, 0, stream>>>(
      attb, w_o, hidden, h1, D_TOK, D_H, D_NH * D_VD);

  // 12) post RMSNorm -> bf16
  rmsnorm_kernel<<<D_TOK, blk, 0, stream>>>(h1, post_ln, yb, D_H, D_H);

  // 13) up = y @ Wup (f32)
  wmma_gemm_kernel<0><<<dim3(D_IM / BN, MG), blk, 0, stream>>>(
      yb, w_up, nullptr, up_f, D_TOK, D_IM, D_H);
  // 14) act = silu(y @ Wgate) * up  -> bf16 (fused epilogue)
  wmma_gemm_kernel<2><<<dim3(D_IM / BN, MG), blk, 0, stream>>>(
      yb, w_gate, up_f, actb, D_TOK, D_IM, D_H);
  // 15) out = act @ Wdown + h1 (residual fused) -> f32 d_out
  wmma_gemm_kernel<3><<<dim3(D_H / BN, MG), blk, 0, stream>>>(
      actb, w_down, h1, (float*)d_out, D_TOK, D_H, D_IM);
}
